// UnifiedDNATransformer_12137577578767
// MI455X (gfx1250) — matrix operand order repaired and checked by reading, not compiled, not run on hardware
//
#include <hip/hip_runtime.h>

#define DEVFN __device__ __forceinline__

typedef __attribute__((ext_vector_type(16))) __bf16 v16bf;
typedef __attribute__((ext_vector_type(8)))  float  v8f;

constexpr int Bn = 4, Sq = 2048, Em = 768, Hh = 12, Dd = 64, FFd = 3072;
constexpr int MROWS = Bn * Sq; // 8192

// ----------------------------------------------------- async-LDS plumbing ---

#if defined(__has_builtin)
#if __has_builtin(__builtin_amdgcn_global_load_async_to_lds_b128) && \
    __has_builtin(__builtin_amdgcn_s_wait_asynccnt)
#define HAVE_ASYNC_LDS 1
#endif
#endif

#ifndef HAVE_ASYNC_LDS
#define HAVE_ASYNC_LDS 0
#endif

#if HAVE_ASYNC_LDS
typedef int v4i_t __attribute__((vector_size(16)));
typedef __attribute__((address_space(1))) v4i_t* gv4i_ptr;   // global int4*
typedef __attribute__((address_space(3))) v4i_t* lv4i_ptr;   // LDS int4*
#define ASYNC_WAIT(n) __builtin_amdgcn_s_wait_asynccnt(n)
#else
#define ASYNC_WAIT(n) ((void)0)
#endif

// copy 16 bytes global -> LDS (async if available)
DEVFN void async_copy16(const unsigned short* g, unsigned short* l) {
#if HAVE_ASYNC_LDS
  __builtin_amdgcn_global_load_async_to_lds_b128((gv4i_ptr)g, (lv4i_ptr)l, 0, 0);
#else
  uint4 d = *(const uint4*)g;
  ((uint2*)l)[0] = make_uint2(d.x, d.y);
  ((uint2*)l)[1] = make_uint2(d.z, d.w);
#endif
}

// ---------------------------------------------------------------- helpers ---

DEVFN unsigned short f2bf(float f) {            // fp32 -> bf16, round-nearest-even
  unsigned int u = __float_as_uint(f);
  u += 0x7FFFu + ((u >> 16) & 1u);
  return (unsigned short)(u >> 16);
}

// A-fragment (16x32 bf16, MxK). ISA: lane L holds row M=L%16;
// lanes 0-15: K = {2v+h}(v<4), {16+2(v-4)+h}; lanes 16-31: +8 per group.
DEVFN v16bf load_a_frag(const unsigned short* lds, int rowOff, int kOff, int stride) {
  const int lane = threadIdx.x & 31;
  const unsigned short* row =
      lds + (rowOff + (lane & 15)) * stride + kOff + ((lane >> 4) << 3);
  union { v16bf v; unsigned int u[8]; } f;
#pragma unroll
  for (int i = 0; i < 4; ++i) f.u[i]     = *(const unsigned int*)(row + 2 * i);
#pragma unroll
  for (int i = 0; i < 4; ++i) f.u[4 + i] = *(const unsigned int*)(row + 16 + 2 * i);
  return f.v;
}

// B-fragment (32x16 bf16, KxN) from LDS stored [n][k].
// ISA: lane L holds col N=L%16; slot t<8 -> K = t + 8*(L>=16), t>=8 -> K = 16 + (t-8) + 8*(L>=16).
DEVFN v16bf load_b_frag(const unsigned short* lds, int nOff, int kOff, int stride) {
  const int lane = threadIdx.x & 31;
  const unsigned short* row =
      lds + (nOff + (lane & 15)) * stride + kOff + ((lane >> 4) << 3);
  union { v16bf v; unsigned int u[8]; } f;
#pragma unroll
  for (int i = 0; i < 4; ++i) f.u[i]     = *(const unsigned int*)(row + 2 * i);
#pragma unroll
  for (int i = 0; i < 4; ++i) f.u[4 + i] = *(const unsigned int*)(row + 16 + 2 * i);
  return f.v;
}

DEVFN v8f wmma_bf16(v16bf a, v16bf b, v8f c) {
  return __builtin_amdgcn_wmma_f32_16x16x32_bf16(false, a, false, b,
                                                 (short)0, c, false, false);
}

// ------------------------------------------------------------ cvt kernel ----

__global__ void cvt_f32_bf16_kernel(const float* __restrict__ in,
                                    unsigned short* __restrict__ out, int n) {
  for (int i = blockIdx.x * blockDim.x + threadIdx.x; i < n;
       i += gridDim.x * blockDim.x)
    out[i] = f2bf(in[i]);
}

// ------------------------------------------------------------ GEMM kernel ---
// C[M,N] = A[M,K](bf16) @ W[N,K]^T(bf16) + bias, templated epilogue.
// Block tile 128x64, BK=64, double-buffered LDS with async global->LDS copies.
// 8 waves: wave (wm=w>>1, wn=w&1) owns rows [32wm,32wm+32) x cols [32wn,32wn+32)
// => 4 independent accumulators, 8 WMMAs per k-slab.

enum { EPI_QKV = 0, EPI_RES = 1, EPI_GELU = 2 };

constexpr int TSTR = 72;  // LDS tile row stride (halfwords): 16B-aligned, no conflicts

DEVFN void gemm_stage(const unsigned short* __restrict__ A,
                      const unsigned short* __restrict__ W,
                      unsigned short* sA, unsigned short* sB,
                      int m0, int n0, int k0, int K, int tid) {
#pragma unroll
  for (int i = 0; i < 4; ++i) {          // A: 128 rows x 8 chunks(16B)
    const int idx = tid + i * 256;
    const int row = idx >> 3, c = (idx & 7) * 8;
    async_copy16(A + (size_t)(m0 + row) * K + k0 + c, sA + row * TSTR + c);
  }
#pragma unroll
  for (int i = 0; i < 2; ++i) {          // W: 64 rows x 8 chunks
    const int idx = tid + i * 256;
    const int row = idx >> 3, c = (idx & 7) * 8;
    async_copy16(W + (size_t)(n0 + row) * K + k0 + c, sB + row * TSTR + c);
  }
}

template <int EPI>
__global__ __launch_bounds__(256) void gemm_bf16_kernel(
    const unsigned short* __restrict__ A, const unsigned short* __restrict__ W,
    const float* __restrict__ bias, const float* __restrict__ res,
    float* __restrict__ outF, unsigned short* __restrict__ outB,
    int M, int N, int K) {
  __shared__ unsigned short sA[2][128 * TSTR];
  __shared__ unsigned short sB[2][64 * TSTR];
  const int tid  = threadIdx.x;
  const int n0   = blockIdx.x * 64;
  const int m0   = blockIdx.y * 128;
  const int w    = tid >> 5;
  const int lane = tid & 31;
  const int wm   = w >> 1;        // 0..3 -> 32-row slice
  const int wn   = w & 1;         // 0..1 -> 32-col slice

  v8f acc[4];
  v8f vz = {};
#pragma unroll
  for (int i = 0; i < 4; ++i) acc[i] = vz;

  const int nk = K >> 6;          // K / 64
  gemm_stage(A, W, sA[0], sB[0], m0, n0, 0, K, tid);

  int buf = 0;
  for (int i = 0; i < nk; ++i) {
    if (i + 1 < nk) {
      gemm_stage(A, W, sA[buf ^ 1], sB[buf ^ 1], m0, n0, (i + 1) << 6, K, tid);
      ASYNC_WAIT(6);              // 6 newest (next slab) may remain outstanding
    } else {
      ASYNC_WAIT(0);
    }
    __syncthreads();              // slab i visible to all waves

    const unsigned short* a_ = sA[buf];
    const unsigned short* b_ = sB[buf];
#pragma unroll
    for (int ks = 0; ks < 2; ++ks) {
      v16bf a0 = load_a_frag(a_, wm * 32,      ks * 32, TSTR);
      v16bf a1 = load_a_frag(a_, wm * 32 + 16, ks * 32, TSTR);
      v16bf b0 = load_b_frag(b_, wn * 32,      ks * 32, TSTR);
      v16bf b1 = load_b_frag(b_, wn * 32 + 16, ks * 32, TSTR);
      acc[0] = wmma_bf16(a0, b0, acc[0]);
      acc[1] = wmma_bf16(a0, b1, acc[1]);
      acc[2] = wmma_bf16(a1, b0, acc[2]);
      acc[3] = wmma_bf16(a1, b1, acc[3]);
    }
    __syncthreads();              // all reads of slab i done before restaging
    buf ^= 1;
  }

  const int hi  = lane >> 4;
  const int ln_ = lane & 15;
#pragma unroll
  for (int mt = 0; mt < 2; ++mt) {
#pragma unroll
    for (int nt = 0; nt < 2; ++nt) {
      v8f a = acc[mt * 2 + nt];
      const int ncol = n0 + wn * 32 + nt * 16 + ln_;
      const float bv = bias[ncol];
#pragma unroll
      for (int r = 0; r < 8; ++r) {
        const int m = m0 + wm * 32 + mt * 16 + r + 8 * hi;
        float v = a[r] + bv;
        if (EPI == EPI_QKV) {        // scatter to [B,H,S,D] bf16
          const int bb = m >> 11, s = m & (Sq - 1);
          const int hh = ncol >> 6, d = ncol & 63;
          outB[(((size_t)(bb * Hh + hh)) * Sq + s) * Dd + d] = f2bf(v);
        } else if (EPI == EPI_RES) { // + residual, fp32
          const size_t idx = (size_t)m * N + ncol;
          outF[idx] = v + res[idx];
        } else {                     // exact GELU, bf16
          const float g = 0.5f * v * (1.0f + erff(v * 0.70710678118f));
          outB[(size_t)m * N + ncol] = f2bf(g);
        }
      }
    }
  }
}

// ------------------------------------------------------ flash attention -----
// One block per (b, h, 128 query rows). 8 waves, wave w owns rows 16w..16w+15.
// Online softmax with ALiBi + key-padding mask; P staged via per-wave LDS rows.

__global__ __launch_bounds__(256) void attn_kernel(
    const unsigned short* __restrict__ Q, const unsigned short* __restrict__ K,
    const unsigned short* __restrict__ V, const unsigned char* __restrict__ mask,
    unsigned short* __restrict__ ctx) {
  __shared__ unsigned short sQ[128 * TSTR];
  __shared__ unsigned short sK[64 * TSTR];
  __shared__ unsigned short sVt[64 * TSTR];   // V transposed: [d][key]
  __shared__ unsigned short sP[128 * TSTR];   // per-wave-private rows

  const int tid  = threadIdx.x;
  const int lane = tid & 31;
  const int w    = tid >> 5;
  const int hi   = lane >> 4;
  const int ln_  = lane & 15;

  const int q0 = blockIdx.x * 128;
  const int h  = blockIdx.y;
  const int b  = blockIdx.z;

  const unsigned short* Qg = Q + ((size_t)(b * Hh + h) * Sq + q0) * Dd;
  const unsigned short* Kg = K + (size_t)(b * Hh + h) * Sq * Dd;
  const unsigned short* Vg = V + (size_t)(b * Hh + h) * Sq * Dd;
  const unsigned char*  mrow = mask + (size_t)b * Sq;

#pragma unroll
  for (int i = 0; i < 4; ++i) {               // stage Q tile 128x64 (async)
    const int idx = tid + i * 256;
    const int row = idx >> 3, c = (idx & 7) * 8;
    async_copy16(Qg + row * 64 + c, sQ + row * TSTR + c);
  }

  const float slope = exp2f(-(float)(h + 1)); // ALiBi slope 1/2^(h+1)

  float mstat[8], lstat[8], scale_[8];
  v8f o[4];
  v8f vz = {};
#pragma unroll
  for (int r = 0; r < 8; ++r) { mstat[r] = -3.0e38f; lstat[r] = 0.0f; }
#pragma unroll
  for (int d = 0; d < 4; ++d) o[d] = vz;

  for (int kb = 0; kb < Sq / 64; ++kb) {
    const int kb0 = kb * 64;
#pragma unroll
    for (int i = 0; i < 2; ++i) {             // stage K row-major (async)
      const int idx = tid + i * 256;
      const int row = idx >> 3, c = (idx & 7) * 8;
      async_copy16(Kg + (size_t)(kb0 + row) * 64 + c, sK + row * TSTR + c);
    }
#pragma unroll
    for (int i = 0; i < 4; ++i) {             // stage V transposed (manual)
      const int idx = tid + i * 256;
      const int row = idx >> 4, c = (idx & 15) * 4;
      uint2 dv = *(const uint2*)(Vg + (size_t)(kb0 + row) * 64 + c);
      sVt[(c + 0) * TSTR + row] = (unsigned short)(dv.x & 0xffffu);
      sVt[(c + 1) * TSTR + row] = (unsigned short)(dv.x >> 16);
      sVt[(c + 2) * TSTR + row] = (unsigned short)(dv.y & 0xffffu);
      sVt[(c + 3) * TSTR + row] = (unsigned short)(dv.y >> 16);
    }
    ASYNC_WAIT(0);                            // K (and first-iter Q) landed
    __syncthreads();

    if (kb + 1 < Sq / 64)
      __builtin_prefetch((const void*)(Vg + (size_t)(kb0 + 64) * 64), 0, 1);

    // S = Q K^T  (contract d=64 in two k=32 WMMA steps)
    v16bf qa0 = load_a_frag(sQ, w * 16, 0, TSTR);
    v16bf qa1 = load_a_frag(sQ, w * 16, 32, TSTR);
    v8f sc[4];
#pragma unroll
    for (int nt = 0; nt < 4; ++nt) {
      v8f c = vz;
      c = wmma_bf16(qa0, load_b_frag(sK, nt * 16, 0, TSTR), c);
      c = wmma_bf16(qa1, load_b_frag(sK, nt * 16, 32, TSTR), c);
      sc[nt] = c;
    }

    float sval[4][8];
#pragma unroll
    for (int nt = 0; nt < 4; ++nt) {
      const int kg = kb0 + nt * 16 + ln_;
      const float mbias = mrow[kg] ? -1.0e30f : 0.0f;
      const float kgf = (float)kg;
#pragma unroll
      for (int r = 0; r < 8; ++r) {
        const float qg = (float)(q0 + w * 16 + r + 8 * hi);
        sval[nt][r] = sc[nt][r] * 0.125f - slope * (qg - kgf) + mbias;
      }
    }

    // online softmax (16-lane row groups; xor<16 stays inside a group)
#pragma unroll
    for (int r = 0; r < 8; ++r) {
      float mx = fmaxf(fmaxf(sval[0][r], sval[1][r]), fmaxf(sval[2][r], sval[3][r]));
#pragma unroll
      for (int off = 1; off < 16; off <<= 1) mx = fmaxf(mx, __shfl_xor(mx, off, 32));
      const float mnew = fmaxf(mstat[r], mx);
      const float scl = __expf(mstat[r] - mnew);
      float rs = 0.0f;
#pragma unroll
      for (int nt = 0; nt < 4; ++nt) {
        const float p = __expf(sval[nt][r] - mnew);
        sval[nt][r] = p;
        rs += p;
      }
#pragma unroll
      for (int off = 1; off < 16; off <<= 1) rs += __shfl_xor(rs, off, 32);
      lstat[r] = lstat[r] * scl + rs;
      mstat[r] = mnew;
      scale_[r] = scl;
    }

    // stage P (own rows only -> no barrier needed), rescale O
#pragma unroll
    for (int nt = 0; nt < 4; ++nt)
#pragma unroll
      for (int r = 0; r < 8; ++r)
        sP[(w * 16 + r + 8 * hi) * TSTR + nt * 16 + ln_] = f2bf(sval[nt][r]);
#pragma unroll
    for (int d = 0; d < 4; ++d)
#pragma unroll
      for (int r = 0; r < 8; ++r) o[d][r] *= scale_[r];

    // O += P V
    v16bf pa0 = load_a_frag(sP, w * 16, 0, TSTR);
    v16bf pa1 = load_a_frag(sP, w * 16, 32, TSTR);
#pragma unroll
    for (int d = 0; d < 4; ++d) {
      o[d] = wmma_bf16(pa0, load_b_frag(sVt, d * 16, 0, TSTR), o[d]);
      o[d] = wmma_bf16(pa1, load_b_frag(sVt, d * 16, 32, TSTR), o[d]);
    }
    __syncthreads();
  }

  // ctx[b, s, h*64+d] = O / l   (bf16, [B,S,E] layout for O-projection)
#pragma unroll
  for (int d = 0; d < 4; ++d)
#pragma unroll
    for (int r = 0; r < 8; ++r) {
      const int qg = q0 + w * 16 + r + 8 * hi;
      const float vout = o[d][r] / lstat[r];
      ctx[((size_t)b * Sq + qg) * Em + h * Dd + d * 16 + ln_] = f2bf(vout);
    }
}

// -------------------------------------------------------------- layernorm ---

template <bool WB>
__global__ __launch_bounds__(256) void layernorm_kernel(
    const float* __restrict__ in, const float* __restrict__ g,
    const float* __restrict__ bb, float* __restrict__ outF,
    unsigned short* __restrict__ outB) {
  __shared__ float r1[256], r2[256];
  const int row = blockIdx.x, tid = threadIdx.x;
  const float* p = in + (size_t)row * Em;
  const float v0 = p[tid], v1 = p[tid + 256], v2 = p[tid + 512];
  r1[tid] = v0 + v1 + v2;
  r2[tid] = v0 * v0 + v1 * v1 + v2 * v2;
  __syncthreads();
  for (int s = 128; s > 0; s >>= 1) {
    if (tid < s) { r1[tid] += r1[tid + s]; r2[tid] += r2[tid + s]; }
    __syncthreads();
  }
  const float mean = r1[0] * (1.0f / 768.0f);
  const float var  = r2[0] * (1.0f / 768.0f) - mean * mean;
  const float rstd = rsqrtf(fmaxf(var, 0.0f) + 1e-5f);
  float* q = outF + (size_t)row * Em;
  unsigned short* qb = WB ? (outB + (size_t)row * Em) : nullptr;
#pragma unroll
  for (int j = 0; j < 3; ++j) {
    const int i = tid + j * 256;
    const float v = (j == 0 ? v0 : (j == 1 ? v1 : v2));
    const float y = (v - mean) * rstd * g[i] + bb[i];
    q[i] = y;
    if (WB) qb[i] = f2bf(y);
  }
}

// ---------------------------------------------------------------- driver ----

extern "C" void kernel_launch(void* const* d_in, const int* in_sizes, int n_in,
                              void* d_out, int out_size, void* d_ws, size_t ws_size,
                              hipStream_t stream) {
  (void)in_sizes; (void)n_in; (void)out_size; (void)ws_size;
  const float* x  = (const float*)d_in[0];
  const unsigned char* mask = (const unsigned char*)d_in[1];
  const float* Wq = (const float*)d_in[2];  const float* bq = (const float*)d_in[3];
  const float* Wk = (const float*)d_in[4];  const float* bk = (const float*)d_in[5];
  const float* Wv = (const float*)d_in[6];  const float* bv = (const float*)d_in[7];
  const float* Wo = (const float*)d_in[8];  const float* bo = (const float*)d_in[9];
  const float* W1 = (const float*)d_in[10]; const float* b1 = (const float*)d_in[11];
  const float* W2 = (const float*)d_in[12]; const float* b2 = (const float*)d_in[13];
  const float* g1 = (const float*)d_in[14]; const float* be1 = (const float*)d_in[15];
  const float* g2 = (const float*)d_in[16]; const float* be2 = (const float*)d_in[17];
  float* out = (float*)d_out;
  char* ws = (char*)d_ws;

  constexpr size_t SZ_Ws  = (size_t)Em * Em * 2;
  constexpr size_t SZ_Wb  = (size_t)FFd * Em * 2;
  constexpr size_t SZ_Abf = (size_t)MROWS * Em * 2;
  constexpr size_t SZ_Af  = (size_t)MROWS * Em * 4;
  constexpr size_t OFF_WQB = 0;
  constexpr size_t OFF_WKB = OFF_WQB + SZ_Ws;
  constexpr size_t OFF_WVB = OFF_WKB + SZ_Ws;
  constexpr size_t OFF_WOB = OFF_WVB + SZ_Ws;
  constexpr size_t OFF_W1B = OFF_WOB + SZ_Ws;
  constexpr size_t OFF_W2B = OFF_W1B + SZ_Wb;
  constexpr size_t OFF_XB  = OFF_W2B + SZ_Wb;         // reused as ctx after QKV
  constexpr size_t OFF_QB  = OFF_XB + SZ_Abf;
  constexpr size_t OFF_KB  = OFF_QB + SZ_Abf;
  constexpr size_t OFF_VB  = OFF_KB + SZ_Abf;
  constexpr size_t OFF_T1  = OFF_VB + SZ_Abf;         // fp32 tmp, reused for FFN2
  constexpr size_t OFF_SRC = OFF_T1 + SZ_Af;
  constexpr size_t OFF_SRCB= OFF_SRC + SZ_Af;
  constexpr size_t OFF_FF1 = OFF_SRCB + SZ_Abf;

  unsigned short* xb  = (unsigned short*)(ws + OFF_XB);
  unsigned short* wqb = (unsigned short*)(ws + OFF_WQB);
  unsigned short* wkb = (unsigned short*)(ws + OFF_WKB);
  unsigned short* wvb = (unsigned short*)(ws + OFF_WVB);
  unsigned short* wob = (unsigned short*)(ws + OFF_WOB);
  unsigned short* w1b = (unsigned short*)(ws + OFF_W1B);
  unsigned short* w2b = (unsigned short*)(ws + OFF_W2B);
  unsigned short* Qb  = (unsigned short*)(ws + OFF_QB);
  unsigned short* Kb  = (unsigned short*)(ws + OFF_KB);
  unsigned short* Vb  = (unsigned short*)(ws + OFF_VB);
  unsigned short* ctx = (unsigned short*)(ws + OFF_XB);
  float*          t1  = (float*)(ws + OFF_T1);
  float*          src = (float*)(ws + OFF_SRC);
  unsigned short* srcb= (unsigned short*)(ws + OFF_SRCB);
  unsigned short* ff1 = (unsigned short*)(ws + OFF_FF1);

  const dim3 blk(256);
  // 1) fp32 -> bf16 conversions
  cvt_f32_bf16_kernel<<<2048, blk, 0, stream>>>(x,  xb,  MROWS * Em);
  cvt_f32_bf16_kernel<<<512,  blk, 0, stream>>>(Wq, wqb, Em * Em);
  cvt_f32_bf16_kernel<<<512,  blk, 0, stream>>>(Wk, wkb, Em * Em);
  cvt_f32_bf16_kernel<<<512,  blk, 0, stream>>>(Wv, wvb, Em * Em);
  cvt_f32_bf16_kernel<<<512,  blk, 0, stream>>>(Wo, wob, Em * Em);
  cvt_f32_bf16_kernel<<<2048, blk, 0, stream>>>(W1, w1b, FFd * Em);
  cvt_f32_bf16_kernel<<<2048, blk, 0, stream>>>(W2, w2b, Em * FFd);

  // 2) QKV projections -> [B,H,S,D] bf16
  const dim3 gE(Em / 64, MROWS / 128);
  gemm_bf16_kernel<EPI_QKV><<<gE, blk, 0, stream>>>(xb, wqb, bq, nullptr, nullptr, Qb, MROWS, Em, Em);
  gemm_bf16_kernel<EPI_QKV><<<gE, blk, 0, stream>>>(xb, wkb, bk, nullptr, nullptr, Kb, MROWS, Em, Em);
  gemm_bf16_kernel<EPI_QKV><<<gE, blk, 0, stream>>>(xb, wvb, bv, nullptr, nullptr, Vb, MROWS, Em, Em);

  // 3) flash attention -> ctx bf16 [B,S,E]
  attn_kernel<<<dim3(Sq / 128, Hh, Bn), blk, 0, stream>>>(Qb, Kb, Vb, mask, ctx);

  // 4) O-projection + residual(x) -> t1 fp32
  gemm_bf16_kernel<EPI_RES><<<gE, blk, 0, stream>>>(ctx, wob, bo, x, t1, nullptr, MROWS, Em, Em);

  // 5) LN1 -> src fp32 + srcb bf16
  layernorm_kernel<true><<<MROWS, blk, 0, stream>>>(t1, g1, be1, src, srcb);

  // 6) FFN1 + GELU -> ff1 bf16
  gemm_bf16_kernel<EPI_GELU><<<dim3(FFd / 64, MROWS / 128), blk, 0, stream>>>(
      srcb, w1b, b1, nullptr, nullptr, ff1, MROWS, FFd, Em);

  // 7) FFN2 + residual(src) -> t1 fp32
  gemm_bf16_kernel<EPI_RES><<<gE, blk, 0, stream>>>(ff1, w2b, b2, src, t1, nullptr, MROWS, Em, FFd);

  // 8) LN2 -> output fp32
  layernorm_kernel<false><<<MROWS, blk, 0, stream>>>(t1, g2, be2, out, nullptr);
}
